// TriangleAttention_32813550141729
// MI455X (gfx1250) — matrix operand order repaired and checked by reading, not compiled, not run on hardware
//
#include <hip/hip_runtime.h>

#define N   384
#define C   128
#define H   4
#define DH  32
#define R   96
#define SD  449
#define MLPH 256

typedef __attribute__((ext_vector_type(16))) __bf16 v16bf;
typedef __attribute__((ext_vector_type(8)))  float  v8f;

union H16 { uint4 q[2]; v16bf v; };

// native f32 -> bf16 (RNE) : let the backend use v_cvt_*bf16* ops
static __device__ __forceinline__ unsigned short f2bf(float f) {
  __bf16 h = (__bf16)f;
  return __builtin_bit_cast(unsigned short, h);
}
static __device__ __forceinline__ float bf2f(unsigned short h) {
  return __uint_as_float(((unsigned int)h) << 16);
}
// 16 contiguous bf16 (B operand: lane n = column, rows koff..koff+15)
static __device__ __forceinline__ v16bf ld16h(const unsigned short* p) {
  H16 h; h.q[0] = *(const uint4*)p; h.q[1] = *(const uint4*)(p + 8); return h.v;
}
// A operand: lane holds row m; K = 16*khalf + idx  -> two 8-half chunks
static __device__ __forceinline__ v16bf ldA16(const unsigned short* rowp, int khalf) {
  H16 h;
  h.q[0] = *(const uint4*)(rowp + 16 * khalf);
  h.q[1] = *(const uint4*)(rowp + 16 * khalf + 8);
  return h.v;
}
static __device__ __forceinline__ v8f wmma_bf16(v16bf a, v16bf b, v8f c) {
  return __builtin_amdgcn_wmma_f32_16x16x32_bf16(false, a, false, b, (short)0, c,
                                                 false, false);
}

// --- CDNA5 async memory->LDS (ASYNCcnt-tracked), per CDNA5 ISA
static __device__ __forceinline__ void async_b128_to_lds(const void* gsrc,
                                                         unsigned int lds_off) {
  asm volatile("global_load_async_to_lds_b128 %0, %1, off"
               :: "v"(lds_off), "v"(gsrc) : "memory");
}
static __device__ __forceinline__ void wait_async0() {
  asm volatile("s_wait_asynccnt 0x0" ::: "memory");
}

// ---------------------------------------------------------------------------
// 1) LayerNorm (f32 in) -> bf16 out. One wave32 per (i,j) vector of 128.
// ---------------------------------------------------------------------------
__global__ __launch_bounds__(256) void ln_bf16_kernel(
    const float* __restrict__ x, const float* __restrict__ gamma,
    const float* __restrict__ beta, unsigned short* __restrict__ xh) {
  int wave = threadIdx.x >> 5, lane = threadIdx.x & 31;
  long long row = (long long)blockIdx.x * 8 + wave;
  const float* xr = x + row * C;
  float4 v = *(const float4*)(xr + lane * 4);
  float s  = v.x + v.y + v.z + v.w;
  float s2 = v.x * v.x + v.y * v.y + v.z * v.z + v.w * v.w;
  #pragma unroll
  for (int m = 1; m < 32; m <<= 1) { s += __shfl_xor(s, m); s2 += __shfl_xor(s2, m); }
  float mean = s * (1.0f / 128.0f);
  float var  = s2 * (1.0f / 128.0f) - mean * mean;
  float rstd = rsqrtf(var + 1e-5f);
  float4 g = *(const float4*)(gamma + lane * 4);
  float4 b = *(const float4*)(beta + lane * 4);
  float y0 = (v.x - mean) * rstd * g.x + b.x;
  float y1 = (v.y - mean) * rstd * g.y + b.y;
  float y2 = (v.z - mean) * rstd * g.z + b.z;
  float y3 = (v.w - mean) * rstd * g.w + b.w;
  uint2 pk;
  pk.x = (unsigned int)f2bf(y0) | ((unsigned int)f2bf(y1) << 16);
  pk.y = (unsigned int)f2bf(y2) | ((unsigned int)f2bf(y3) << 16);
  *(uint2*)(xh + row * C + lane * 4) = pk;
}

// ---------------------------------------------------------------------------
// 2) Mean over one axis of xh: out[b*C+c] = mean_t xh[b*so + t*si + c]
// ---------------------------------------------------------------------------
__global__ __launch_bounds__(128) void meanreduce_kernel(
    const unsigned short* __restrict__ xh, float* __restrict__ out,
    long long so, long long si) {
  int b = blockIdx.x, c = threadIdx.x;
  const unsigned short* p = xh + (long long)b * so + c;
  float acc = 0.f;
  for (int t = 0; t < N; ++t) acc += bf2f(p[(long long)t * si]);
  out[b * C + c] = acc * (1.0f / (float)N);
}

// ---------------------------------------------------------------------------
// 3) Bias MLP stage: out[n,:] = act( concat(A[n],B[n]) @ W )   (f32, tiny)
// ---------------------------------------------------------------------------
__global__ __launch_bounds__(256) void mlp_kernel(
    const float* __restrict__ A, int wa, const float* __restrict__ B, int wb,
    const float* __restrict__ W, int ncols, float* __restrict__ out, int act) {
  __shared__ float sin_[SD + C];
  int n = blockIdx.x, tid = threadIdx.x, K = wa + wb;
  for (int t = tid; t < K; t += blockDim.x)
    sin_[t] = (t < wa) ? A[(long long)n * wa + t] : B[(long long)n * wb + (t - wa)];
  __syncthreads();
  for (int col = tid; col < ncols; col += blockDim.x) {
    float acc = 0.f;
    for (int k = 0; k < K; ++k) acc += sin_[k] * W[(long long)k * ncols + col];
    out[(long long)n * ncols + col] = act ? tanhf(acc) : acc;
  }
}

// ---------------------------------------------------------------------------
// 4) Low-rank bias: bm[h,q,k] = sum_r qb[q, h*R+r] * kb[k, h*R+r]
// ---------------------------------------------------------------------------
__global__ __launch_bounds__(256) void biasouter_kernel(
    const float* __restrict__ qb, const float* __restrict__ kb,
    float* __restrict__ bm) {
  int h = blockIdx.y;
  int lin = blockIdx.x * blockDim.x + threadIdx.x;   // < N*N
  int q = lin / N, k = lin % N;
  const float* qp = qb + (long long)q * (H * R) + h * R;
  const float* kp = kb + (long long)k * (H * R) + h * R;
  float acc = 0.f;
  #pragma unroll 8
  for (int r = 0; r < R; ++r) acc += qp[r] * kp[r];
  bm[((long long)h * N + q) * N + k] = acc;
}

// ---------------------------------------------------------------------------
// 5) Fused per-(row i, head h) attention. One workgroup = one (i,h).
//    LDS-resident: X row (bf16, async-staged), W slices (transposed),
//    Q/K (row-major), V^T, per-wave P scratch.  v_wmma_f32_16x16x32_bf16.
// ---------------------------------------------------------------------------
__global__ __launch_bounds__(512) void attn_kernel(
    const unsigned short* __restrict__ xh,
    const float* __restrict__ Wq, const float* __restrict__ Wk,
    const float* __restrict__ Wv,
    const float* __restrict__ bm, unsigned short* __restrict__ obuf) {
  extern __shared__ unsigned short smem[];
  unsigned short* sX  = smem;                  // [384][128]
  unsigned short* sWT = sX + N * C;            // 3 x [32][128]  (W^T per head)
  unsigned short* sQ  = sWT + 3 * DH * C;      // [384][32]  (pre-scaled)
  unsigned short* sK  = sQ + N * DH;           // [384][32]
  unsigned short* sVt = sK + N * DH;           // [32][384]
  unsigned short* sP  = sVt + DH * N;          // nwaves x [16][32]

  const int h = blockIdx.x, i = blockIdx.y;
  const int tid = threadIdx.x, lane = tid & 31, wave = tid >> 5;
  const int nwaves = blockDim.x >> 5;
  const int m = lane & 15, khalf = lane >> 4, koff = khalf * 16;

  // ---- stage X row into LDS via async DMA; weights via transpose copy ----
  {
    const char* gx = (const char*)(xh + (long long)i * N * C);
    unsigned int xoff = (unsigned int)((char*)sX - (char*)smem);   // == 0
    for (int t = tid; t < (N * C) / 8; t += blockDim.x)
      async_b128_to_lds(gx + (size_t)t * 16, xoff + t * 16);
    for (int t = tid; t < 3 * DH * C; t += blockDim.x) {
      int mat = t / (DH * C), r = t % (DH * C);
      int d = r >> 7, c = r & 127;
      const float* ws = (mat == 0) ? Wq : ((mat == 1) ? Wk : Wv);
      sWT[t] = f2bf(ws[(long long)c * (H * DH) + h * DH + d]);
    }
    wait_async0();
  }
  __syncthreads();

  // ---- projections (three straight-line passes: Q, K, V^T) ----
  for (int job = wave; job < 48; job += nwaves) {          // Q, pre-scaled
    int mt = job >> 1, nt = job & 1;
    v8f acc = {};
    #pragma unroll
    for (int kc = 0; kc < 4; ++kc) {
      v16bf a = ldA16(sX + (mt * 16 + m) * C + kc * 32, khalf);
      v16bf b = ld16h(sWT + (nt * 16 + m) * C + kc * 32 + koff);
      acc = wmma_bf16(a, b, acc);
    }
    #pragma unroll
    for (int r = 0; r < 8; ++r)
      sQ[(mt * 16 + r + 8 * khalf) * DH + nt * 16 + m] =
          f2bf(acc[r] * 0.1767766953f);                    // 1/sqrt(32)
  }
  for (int job = wave; job < 48; job += nwaves) {          // K
    int mt = job >> 1, nt = job & 1;
    const unsigned short* wt = sWT + DH * C;
    v8f acc = {};
    #pragma unroll
    for (int kc = 0; kc < 4; ++kc) {
      v16bf a = ldA16(sX + (mt * 16 + m) * C + kc * 32, khalf);
      v16bf b = ld16h(wt + (nt * 16 + m) * C + kc * 32 + koff);
      acc = wmma_bf16(a, b, acc);
    }
    #pragma unroll
    for (int r = 0; r < 8; ++r)
      sK[(mt * 16 + r + 8 * khalf) * DH + nt * 16 + m] = f2bf(acc[r]);
  }
  for (int job = wave; job < 48; job += nwaves) {          // V (stored transposed)
    int mt = job >> 1, nt = job & 1;
    const unsigned short* wt = sWT + 2 * DH * C;
    v8f acc = {};
    #pragma unroll
    for (int kc = 0; kc < 4; ++kc) {
      v16bf a = ldA16(sX + (mt * 16 + m) * C + kc * 32, khalf);
      v16bf b = ld16h(wt + (nt * 16 + m) * C + kc * 32 + koff);
      acc = wmma_bf16(a, b, acc);
    }
    #pragma unroll
    for (int r = 0; r < 8; ++r)
      sVt[(nt * 16 + m) * N + mt * 16 + r + 8 * khalf] = f2bf(acc[r]);
  }
  __syncthreads();

  // ---- attention: per 16-row q tile, exact two-pass softmax ----
  unsigned short* sPw = sP + wave * 512;
  const float* bmh = bm + (long long)h * N * N;
  for (int qt = wave; qt < N / 16; qt += nwaves) {
    {   // keep this q-tile's bias rows warm (global_prefetch_b8)
      int prow = qt * 16 + lane; if (prow > N - 1) prow = N - 1;
      __builtin_prefetch(bmh + (long long)prow * N, 0, 1);
    }
    v16bf aq = ldA16(sQ + (qt * 16 + m) * DH, khalf);
    float sc[24][8];
    #pragma unroll
    for (int kt = 0; kt < 24; ++kt) {
      v16bf bk = ld16h(sK + (kt * 16 + m) * DH + koff);
      v8f c = {};
      c = wmma_bf16(aq, bk, c);
      #pragma unroll
      for (int r = 0; r < 8; ++r)
        sc[kt][r] = c[r] + bmh[(long long)(qt * 16 + r + 8 * khalf) * N + kt * 16 + m];
    }
    float mx[8], l[8];
    #pragma unroll
    for (int r = 0; r < 8; ++r) mx[r] = -3.0e38f;
    #pragma unroll
    for (int kt = 0; kt < 24; ++kt)
      #pragma unroll
      for (int r = 0; r < 8; ++r) mx[r] = fmaxf(mx[r], sc[kt][r]);
    #pragma unroll
    for (int r = 0; r < 8; ++r) {
      #pragma unroll
      for (int msk = 1; msk < 16; msk <<= 1)
        mx[r] = fmaxf(mx[r], __shfl_xor(mx[r], msk));   // reduce within 16-lane half
      l[r] = 0.f;
    }
    #pragma unroll
    for (int kt = 0; kt < 24; ++kt)
      #pragma unroll
      for (int r = 0; r < 8; ++r) {
        float pv = __expf(sc[kt][r] - mx[r]);
        sc[kt][r] = pv; l[r] += pv;
      }
    #pragma unroll
    for (int r = 0; r < 8; ++r)
      #pragma unroll
      for (int msk = 1; msk < 16; msk <<= 1) l[r] += __shfl_xor(l[r], msk);

    // AV: pairs of k-tiles -> A = [P_t | P_t+1] via wave-local LDS relayout
    v8f o0 = {}, o1 = {};
    for (int u = 0; u < 12; ++u) {
      #pragma unroll
      for (int t = 0; t < 2; ++t)
        #pragma unroll
        for (int r = 0; r < 8; ++r)
          sPw[(r + 8 * khalf) * 32 + t * 16 + m] = f2bf(sc[2 * u + t][r]);
      v16bf ap  = ldA16(sPw + m * 32, khalf);
      v16bf bv0 = ld16h(sVt + (long long)m * N + u * 32 + koff);
      o0 = wmma_bf16(ap, bv0, o0);
      v16bf bv1 = ld16h(sVt + (long long)(16 + m) * N + u * 32 + koff);
      o1 = wmma_bf16(ap, bv1, o1);
    }
    #pragma unroll
    for (int r = 0; r < 8; ++r) {
      float inv = 1.0f / l[r];
      int row = qt * 16 + r + 8 * khalf;
      long long base = ((long long)i * N + row) * C + h * DH;
      obuf[base + m]      = f2bf(o0[r] * inv);
      obuf[base + 16 + m] = f2bf(o1[r] * inv);
    }
  }
}

// ---------------------------------------------------------------------------
// 6) Gate + output projection: out = (sigmoid(X@Wg+bg) * O) @ Wo + bo
//    128 flattened rows per block, async-staged LDS tiles, WMMA both GEMMs.
// ---------------------------------------------------------------------------
__global__ __launch_bounds__(256) void gateout_kernel(
    const unsigned short* __restrict__ xh, const unsigned short* __restrict__ obuf,
    const float* __restrict__ Wg, const float* __restrict__ bg,
    const float* __restrict__ Wo, const float* __restrict__ bo,
    float* __restrict__ out) {
  extern __shared__ unsigned short smem[];
  unsigned short* sX   = smem;              // [128][128]
  unsigned short* sO   = sX + 128 * C;
  unsigned short* sG   = sO + 128 * C;
  unsigned short* sWgT = sG + 128 * C;      // [128][128] W^T
  unsigned short* sWoT = sWgT + C * C;
  const int tid = threadIdx.x, lane = tid & 31, wave = tid >> 5;
  const int nwaves = blockDim.x >> 5;
  const int m = lane & 15, khalf = lane >> 4, koff = khalf * 16;
  long long rowbase = (long long)blockIdx.x * 128;

  {
    const char* gx = (const char*)(xh + rowbase * C);
    const char* go = (const char*)(obuf + rowbase * C);
    unsigned int xoff = (unsigned int)((char*)sX - (char*)smem);   // 0
    unsigned int ooff = (unsigned int)((char*)sO - (char*)smem);   // 32768
    for (int t = tid; t < (128 * C) / 8; t += blockDim.x) {
      async_b128_to_lds(gx + (size_t)t * 16, xoff + t * 16);
      async_b128_to_lds(go + (size_t)t * 16, ooff + t * 16);
    }
    for (int t = tid; t < C * C; t += blockDim.x) {
      int d = t >> 7, c = t & 127;
      sWgT[t] = f2bf(Wg[(long long)c * C + d]);
      sWoT[t] = f2bf(Wo[(long long)c * C + d]);
    }
    wait_async0();
  }
  __syncthreads();

  for (int job = wave; job < 64; job += nwaves) {
    int mt = job >> 3, nt = job & 7;
    v8f acc = {};
    #pragma unroll
    for (int kc = 0; kc < 4; ++kc) {
      v16bf a = ldA16(sX + (mt * 16 + m) * C + kc * 32, khalf);
      v16bf b = ld16h(sWgT + (nt * 16 + m) * C + kc * 32 + koff);
      acc = wmma_bf16(a, b, acc);
    }
    #pragma unroll
    for (int r = 0; r < 8; ++r) {
      int row = mt * 16 + r + 8 * khalf, col = nt * 16 + m;
      float gv = 1.0f / (1.0f + __expf(-(acc[r] + bg[col])));
      sG[row * C + col] = f2bf(gv * bf2f(sO[row * C + col]));
    }
  }
  __syncthreads();

  for (int job = wave; job < 64; job += nwaves) {
    int mt = job >> 3, nt = job & 7;
    v8f acc = {};
    #pragma unroll
    for (int kc = 0; kc < 4; ++kc) {
      v16bf a = ldA16(sG + (mt * 16 + m) * C + kc * 32, khalf);
      v16bf b = ld16h(sWoT + (nt * 16 + m) * C + kc * 32 + koff);
      acc = wmma_bf16(a, b, acc);
    }
    #pragma unroll
    for (int r = 0; r < 8; ++r) {
      int row = mt * 16 + r + 8 * khalf, col = nt * 16 + m;
      out[(rowbase + row) * C + col] = acc[r] + bo[col];
    }
  }
}

// ---------------------------------------------------------------------------
extern "C" void kernel_launch(void* const* d_in, const int* in_sizes, int n_in,
                              void* d_out, int out_size, void* d_ws, size_t ws_size,
                              hipStream_t stream) {
  const float* x    = (const float*)d_in[0];
  const float* sinp = (const float*)d_in[1];
  const float* lng  = (const float*)d_in[2];
  const float* lnb  = (const float*)d_in[3];
  const float* Wq1  = (const float*)d_in[4];
  const float* Wq2  = (const float*)d_in[5];
  const float* Wq3  = (const float*)d_in[6];
  const float* Wk1  = (const float*)d_in[7];
  const float* Wk2  = (const float*)d_in[8];
  const float* Wk3  = (const float*)d_in[9];
  const float* Wqa  = (const float*)d_in[10];
  const float* Wka  = (const float*)d_in[11];
  const float* Wva  = (const float*)d_in[12];
  const float* Wg   = (const float*)d_in[13];
  const float* bg   = (const float*)d_in[14];
  const float* Wo   = (const float*)d_in[15];
  const float* bo   = (const float*)d_in[16];
  float* out = (float*)d_out;

  char* p = (char*)d_ws;
  auto carve = [&](size_t bytes) { char* r = p; p += (bytes + 255) & ~(size_t)255; return (void*)r; };
  unsigned short* xhb  = (unsigned short*)carve((size_t)N * N * C * 2);
  unsigned short* obuf = (unsigned short*)carve((size_t)N * N * C * 2);
  float* rowmean = (float*)carve((size_t)N * C * 4);
  float* colmean = (float*)carve((size_t)N * C * 4);
  float* h1  = (float*)carve((size_t)N * MLPH * 4);
  float* h2  = (float*)carve((size_t)N * MLPH * 4);
  float* obq = (float*)carve((size_t)N * H * R * 4);
  float* obk = (float*)carve((size_t)N * H * R * 4);
  float* bmx = (float*)carve((size_t)H * N * N * 4);

  ln_bf16_kernel<<<dim3(N * N / 8), dim3(256), 0, stream>>>(x, lng, lnb, xhb);
  meanreduce_kernel<<<dim3(N), dim3(C), 0, stream>>>(xhb, rowmean, (long long)N * C, (long long)C);
  meanreduce_kernel<<<dim3(N), dim3(C), 0, stream>>>(xhb, colmean, (long long)C, (long long)N * C);

  mlp_kernel<<<dim3(N), dim3(256), 0, stream>>>(rowmean, C, sinp, SD, Wq1, MLPH, h1, 1);
  mlp_kernel<<<dim3(N), dim3(256), 0, stream>>>(h1, MLPH, nullptr, 0, Wq2, MLPH, h2, 1);
  mlp_kernel<<<dim3(N), dim3(256), 0, stream>>>(h2, MLPH, nullptr, 0, Wq3, H * R, obq, 0);
  mlp_kernel<<<dim3(N), dim3(256), 0, stream>>>(colmean, C, sinp, SD, Wk1, MLPH, h1, 1);
  mlp_kernel<<<dim3(N), dim3(256), 0, stream>>>(h1, MLPH, nullptr, 0, Wk2, MLPH, h2, 1);
  mlp_kernel<<<dim3(N), dim3(256), 0, stream>>>(h2, MLPH, nullptr, 0, Wk3, H * R, obk, 0);

  biasouter_kernel<<<dim3(N * N / 256, H), dim3(256), 0, stream>>>(obq, obk, bmx);

  size_t attn_lds = (size_t)(N * C + 3 * DH * C + 2 * N * DH + DH * N + 16 * 512) * 2;
  attn_kernel<<<dim3(H, N), dim3(512), attn_lds, stream>>>(xhb, Wqa, Wka, Wva, bmx, obuf);

  size_t go_lds = (size_t)(3 * 128 * C + 2 * C * C) * 2;
  gateout_kernel<<<dim3(N * N / 128), dim3(256), go_lds, stream>>>(xhb, obuf, Wg, bg, Wo, bo, out);
}
